// SelfAttention_7095285973178
// MI455X (gfx1250) — matrix operand order repaired and checked by reading, not compiled, not run on hardware
//
#include <hip/hip_runtime.h>
#include <hip/hip_bf16.h>

// ---------------------------------------------------------------------------
// Self-attention (B=4, T=2048, C=1024, H=16, D=64) on MI455X (gfx1250).
// All GEMMs via v_wmma_f32_16x16x32_f16 (f16 in, f32 accumulate).
// Softmax in f32 with online (flash) accumulation.
// Operand staging via async global->LDS copies (ASYNCcnt), double-buffered
// and shared across the 8 waves of each workgroup.
// ---------------------------------------------------------------------------

#define BATCH 4
#define SEQ   2048
#define CH    1024
#define NHEAD 16
#define HDIM  64
#define NTOK  (BATCH * SEQ)        // 8192
#define N3C   (3 * CH)             // 3072

typedef __attribute__((ext_vector_type(16))) _Float16 v16h;
typedef __attribute__((ext_vector_type(8)))  _Float16 v8h;
typedef __attribute__((ext_vector_type(8)))  float    v8f;
typedef int v4i_vs __attribute__((vector_size(16)));   // matches builtin param

union V16 { v16h v; v8h h[2]; };

static __device__ __forceinline__ v8f wmma_f16(v16h a, v16h b, v8f c) {
  // D = A(16x32 f16) * B(32x16 f16) + C(16x16 f32)
  return __builtin_amdgcn_wmma_f32_16x16x32_f16(
      /*neg_a=*/false, a, /*neg_b=*/false, b,
      /*c_mod=*/(short)0, c, /*reuse_a=*/false, /*reuse_b=*/false);
}

static __device__ __forceinline__ v8f zero8() {
  v8f z = {0.f, 0.f, 0.f, 0.f, 0.f, 0.f, 0.f, 0.f};
  return z;
}

// ---------------------------------------------------------------------------
// Async global->LDS 16B copy (gfx1250 GLOBAL_LOAD_ASYNC_TO_LDS_B128).
// Param 0 is v4i addrspace(1)*, param 1 is the LDS address (addrspace(3)).
// Falls back to a synchronous load+DS-store if the builtin is unavailable.
// ---------------------------------------------------------------------------
#if defined(__has_builtin)
#if __has_builtin(__builtin_amdgcn_global_load_async_to_lds_b128)
#define HAVE_ASYNC_LDS 1
#endif
#endif
#ifndef HAVE_ASYNC_LDS
#define HAVE_ASYNC_LDS 0
#endif

static __device__ __forceinline__ void g2l_b128(const _Float16* g, _Float16* l) {
#if HAVE_ASYNC_LDS
  typedef __attribute__((address_space(1))) v4i_vs* gp_t;
  typedef __attribute__((address_space(3))) v4i_vs* lp_t;
  gp_t gp = (gp_t)(__attribute__((address_space(1))) void*)g;
  lp_t lp = (lp_t)(__attribute__((address_space(3))) void*)l;
  __builtin_amdgcn_global_load_async_to_lds_b128(gp, lp, /*imm offset=*/0,
                                                 /*cpol=*/0);
#else
  *(v8h*)l = *(const v8h*)g;
#endif
}

template <int N>
static __device__ __forceinline__ void wait_async() {
#if HAVE_ASYNC_LDS
#if __has_builtin(__builtin_amdgcn_s_wait_asynccnt)
  __builtin_amdgcn_s_wait_asynccnt(N);
#else
  asm volatile("s_wait_asynccnt %0" ::"n"(N));
#endif
#endif
}

// ---------------------------------------------------------------------------
// f32 -> f16 elementwise convert
// ---------------------------------------------------------------------------
__global__ void cvt_f32_to_f16(const float* __restrict__ src,
                               _Float16* __restrict__ dst, int n) {
  int i = blockIdx.x * blockDim.x + threadIdx.x;
  if (i < n) dst[i] = (_Float16)src[i];
}

// f32 [K][N] -> f16 [N][K] (transpose so WMMA B-operand loads are contiguous)
__global__ void cvt_f32_to_f16_T(const float* __restrict__ src,
                                 _Float16* __restrict__ dst, int K, int N) {
  int i = blockIdx.x * blockDim.x + threadIdx.x;  // output-linear: i = n*K + k
  if (i < K * N) {
    int n = i / K;
    int k = i - n * K;
    dst[i] = (_Float16)src[(size_t)k * N + n];
  }
}

// ---------------------------------------------------------------------------
// QKV GEMM: qkv = xh[NTOK,CH] @ waT[N3C,CH]^T + b_attn
// Block = 8 waves sharing one 64-wide N-tile; B tile staged per K-step into
// LDS via async copy (double-buffered). Each wave computes a 32x64 tile
// (2x4 of 16x16), 8 WMMAs per 32-deep K-step.
// Writes q,k as f16 [B,H,T,D]; writes v transposed as f16 [B,H,D,T].
// ---------------------------------------------------------------------------
__global__ __launch_bounds__(256) void qkv_gemm(
    const _Float16* __restrict__ xh, const _Float16* __restrict__ waT,
    const float* __restrict__ b_attn, _Float16* __restrict__ qh,
    _Float16* __restrict__ kh, _Float16* __restrict__ vTh) {
  __shared__ __align__(64) _Float16 sb[2][64 * 32];  // [col][k] f16, 4KB x2

  const int tid  = threadIdx.x;
  const int wv   = tid >> 5;
  const int lane = tid & 31;
  const int row  = lane & 15;
  const int hi   = lane >> 4;
  const int tilesN = N3C / 64;                    // 48
  const int tn = blockIdx.x % tilesN;             // shared by the whole block
  const int tm = (blockIdx.x / tilesN) * 8 + wv;  // 0..255

  const _Float16* wt = waT + (size_t)tn * 64 * CH;  // this block's B panel
  const int bcol = tid >> 2;           // 0..63 : one column per 4 threads
  const int bofs = (tid & 3) * 8;      // 16B chunk within the 32-half column

  v8f acc[2][4];
#pragma unroll
  for (int i = 0; i < 2; i++)
#pragma unroll
    for (int j = 0; j < 4; j++) acc[i][j] = zero8();

  // preload K-step 0 into buffer 0
  g2l_b128(wt + (size_t)bcol * CH + bofs, sb[0] + bcol * 32 + bofs);

  for (int kb = 0; kb < CH; kb += 32) {
    const int cur = (kb >> 5) & 1;
    const int kbn = (kb + 32) & (CH - 1);  // wrap: harmless dead prefetch at end
    g2l_b128(wt + (size_t)bcol * CH + kbn + bofs, sb[cur ^ 1] + bcol * 32 + bofs);
    wait_async<1>();
    __syncthreads();

    V16 a[2];
#pragma unroll
    for (int i = 0; i < 2; i++) {
      const _Float16* ar = xh + (size_t)(tm * 32 + i * 16 + row) * CH + kb;
      a[i].h[0] = *(const v8h*)(ar + hi * 16);         // K {0..7} / {16..23}
      a[i].h[1] = *(const v8h*)(ar + hi * 16 + 8);     // K {8..15} / {24..31}
      __builtin_prefetch(ar + 64);
    }
    const _Float16* sbc = sb[cur];
#pragma unroll
    for (int j = 0; j < 4; j++) {
      v16h bb = *(const v16h*)(sbc + (j * 16 + row) * 32 + hi * 16);
      acc[0][j] = wmma_f16(a[0].v, bb, acc[0][j]);
      acc[1][j] = wmma_f16(a[1].v, bb, acc[1][j]);
    }
    __syncthreads();
  }

  // Epilogue: bias add, convert, scatter into q/k/vT layouts.
#pragma unroll
  for (int i = 0; i < 2; i++) {
#pragma unroll
    for (int j = 0; j < 4; j++) {
      const int n    = tn * 64 + j * 16 + row;   // global output column
      const float bs = b_attn[n];
      const int part = n >> 10;                  // 0=q 1=k 2=v
      const int c    = n & (CH - 1);
      const int h    = c >> 6;                   // head
      const int d    = c & (HDIM - 1);
#pragma unroll
      for (int vv = 0; vv < 8; vv++) {
        const int tok = tm * 32 + i * 16 + vv + hi * 8;  // C-layout row
        const int b   = tok >> 11;
        const int t   = tok & (SEQ - 1);
        const _Float16 hv = (_Float16)(acc[i][j][vv] + bs);
        const size_t bh = (size_t)(b * NHEAD + h);
        if (part == 0)      qh[(bh * SEQ + t) * HDIM + d]  = hv;
        else if (part == 1) kh[(bh * SEQ + t) * HDIM + d]  = hv;
        else                vTh[(bh * HDIM + d) * SEQ + t] = hv;
      }
    }
  }
}

// ---------------------------------------------------------------------------
// Flash attention: block = 8 waves = 8 consecutive q-tiles of one (b,h).
// K/V blocks (32 keys) staged into LDS via async copy, double-buffered,
// shared by all 8 waves (8x cut in K/V global traffic).
// Per 32-key iteration: 4 WMMAs for S = q k^T, f32 online softmax,
// P staged via wave-private LDS (C-layout -> A-layout), 4 WMMAs for O += P V.
// ---------------------------------------------------------------------------
__global__ __launch_bounds__(256) void flash_attn(
    const _Float16* __restrict__ q, const _Float16* __restrict__ k,
    const _Float16* __restrict__ vT, _Float16* __restrict__ yh) {
  __shared__ __align__(64) _Float16 sk[2][32 * 64];  // [key][d]   4KB x2
  __shared__ __align__(64) _Float16 sv[2][64 * 32];  // [d][key]   4KB x2
  __shared__ __align__(64) _Float16 sp[8][16 * 32];  // per-wave P staging

  const int tid  = threadIdx.x;
  const int wv   = tid >> 5;
  const int lane = tid & 31;
  const int row  = lane & 15;
  const int hi   = lane >> 4;
  const int gw   = blockIdx.x * 8 + wv;     // 8192 waves
  const int bh   = gw >> 7;                 // (b*H + h); 8 | 128 so uniform/block
  const int qt   = gw & 127;
  const int b    = bh >> 4;
  const int h    = bh & 15;

  const _Float16* kbase = k + (size_t)bh * SEQ * HDIM;   // [key][d], rows 128B
  const _Float16* vbase = vT + (size_t)bh * HDIM * SEQ;  // [d][key]

  // cooperative fill indices
  const int vrow = tid >> 2;            // 0..63
  const int vofs = (tid & 3) * 8;       // 16B chunk

  // Q tile in A-layout, split into two 32-deep K halves (d 0..31 / 32..63).
  const _Float16* qr = q + ((size_t)bh * SEQ + qt * 16 + row) * HDIM;
  V16 aq0, aq1;
  aq0.h[0] = *(const v8h*)(qr + hi * 16);
  aq0.h[1] = *(const v8h*)(qr + hi * 16 + 8);
  aq1.h[0] = *(const v8h*)(qr + 32 + hi * 16);
  aq1.h[1] = *(const v8h*)(qr + 32 + hi * 16 + 8);

  float m[8], lsum[8];
  v8f acc[4];
#pragma unroll
  for (int i = 0; i < 8; i++) { m[i] = -1e30f; lsum[i] = 0.f; }
#pragma unroll
  for (int n = 0; n < 4; n++) acc[n] = zero8();

  _Float16* pl = sp[wv];

  // preload key-block 0: K block is a fully contiguous 4KB span
  g2l_b128(kbase + tid * 8, sk[0] + tid * 8);
  g2l_b128(vbase + (size_t)vrow * SEQ + vofs, sv[0] + vrow * 32 + vofs);

  for (int kb = 0; kb < SEQ; kb += 32) {
    const int cur = (kb >> 5) & 1;
    const int kbn = (kb + 32) & (SEQ - 1);  // wrap: harmless dead prefetch
    g2l_b128(kbase + kbn * HDIM + tid * 8, sk[cur ^ 1] + tid * 8);
    g2l_b128(vbase + (size_t)vrow * SEQ + kbn + vofs, sv[cur ^ 1] + vrow * 32 + vofs);
    wait_async<2>();
    __syncthreads();

    const _Float16* skc = sk[cur];
    const _Float16* svc = sv[cur];

    // ---- S = (q k^T) * 1/sqrt(D) for two 16-key sub-blocks ----
    v8f s0 = zero8(), s1 = zero8();
    {
      const _Float16* kr0 = skc + (size_t)row * HDIM;
      v16h b00 = *(const v16h*)(kr0 + hi * 16);        // d 0..31
      v16h b01 = *(const v16h*)(kr0 + 32 + hi * 16);   // d 32..63
      s0 = wmma_f16(aq0.v, b00, s0);
      s0 = wmma_f16(aq1.v, b01, s0);
      const _Float16* kr1 = skc + (size_t)(16 + row) * HDIM;
      v16h b10 = *(const v16h*)(kr1 + hi * 16);
      v16h b11 = *(const v16h*)(kr1 + 32 + hi * 16);
      s1 = wmma_f16(aq0.v, b10, s1);
      s1 = wmma_f16(aq1.v, b11, s1);
    }

    // ---- online softmax (f32). Row r = vv + 8*hi lives on a 16-lane group.
#pragma unroll
    for (int vv = 0; vv < 8; vv++) {
      float a0 = s0[vv] * 0.125f;   // 1/sqrt(64)
      float a1 = s1[vv] * 0.125f;
      float mx = fmaxf(a0, a1);
      mx = fmaxf(mx, __shfl_xor(mx, 1, 32));
      mx = fmaxf(mx, __shfl_xor(mx, 2, 32));
      mx = fmaxf(mx, __shfl_xor(mx, 4, 32));
      mx = fmaxf(mx, __shfl_xor(mx, 8, 32));
      const float mn = fmaxf(m[vv], mx);
      const float cr = __expf(m[vv] - mn);
      m[vv] = mn;
      const float p0 = __expf(a0 - mn);
      const float p1 = __expf(a1 - mn);
      s0[vv] = p0;
      s1[vv] = p1;
      float rs = p0 + p1;
      rs += __shfl_xor(rs, 1, 32);
      rs += __shfl_xor(rs, 2, 32);
      rs += __shfl_xor(rs, 4, 32);
      rs += __shfl_xor(rs, 8, 32);
      lsum[vv] = lsum[vv] * cr + rs;
#pragma unroll
      for (int n = 0; n < 4; n++) acc[n][vv] *= cr;
    }

    // ---- P: C-layout regs -> LDS (row-major [16 rows][32 keys]) ----
#pragma unroll
    for (int vv = 0; vv < 8; vv++) {
      const int r = vv + hi * 8;
      pl[r * 32 + row]      = (_Float16)s0[vv];
      pl[r * 32 + 16 + row] = (_Float16)s1[vv];
    }
    asm volatile("" ::: "memory");  // keep DS store->load order (same wave, in-order LDS)

    // ---- reload P in A-layout ----
    V16 aP;
    aP.h[0] = *(const v8h*)(pl + row * 32 + hi * 16);
    aP.h[1] = *(const v8h*)(pl + row * 32 + hi * 16 + 8);

    // ---- O += P V : sv rows are contiguous along keys ----
#pragma unroll
    for (int n = 0; n < 4; n++) {
      v16h bv = *(const v16h*)(svc + (size_t)(n * 16 + row) * 32 + hi * 16);
      acc[n] = wmma_f16(aP.v, bv, acc[n]);
    }
    __syncthreads();
  }

  // ---- normalize and store attention output as f16 [B,T,C] ----
#pragma unroll
  for (int vv = 0; vv < 8; vv++) {
    const float inv = 1.0f / lsum[vv];
    const int tok = qt * 16 + vv + hi * 8;
    const size_t orow = ((size_t)b * SEQ + tok) * CH + h * HDIM;
#pragma unroll
    for (int n = 0; n < 4; n++) {
      yh[orow + n * 16 + row] = (_Float16)(acc[n][vv] * inv);
    }
  }
}

// ---------------------------------------------------------------------------
// Projection GEMM: out = yh[NTOK,CH] @ wpT[CH,CH]^T + b_proj (f32 out)
// Same blocked structure as qkv_gemm (shared B panel staged via async LDS).
// ---------------------------------------------------------------------------
__global__ __launch_bounds__(256) void proj_gemm(
    const _Float16* __restrict__ yh, const _Float16* __restrict__ wpT,
    const float* __restrict__ b_proj, float* __restrict__ out) {
  __shared__ __align__(64) _Float16 sb[2][64 * 32];

  const int tid  = threadIdx.x;
  const int wv   = tid >> 5;
  const int lane = tid & 31;
  const int row  = lane & 15;
  const int hi   = lane >> 4;
  const int tilesN = CH / 64;                     // 16
  const int tn = blockIdx.x % tilesN;
  const int tm = (blockIdx.x / tilesN) * 8 + wv;  // 0..255

  const _Float16* wt = wpT + (size_t)tn * 64 * CH;
  const int bcol = tid >> 2;
  const int bofs = (tid & 3) * 8;

  v8f acc[2][4];
#pragma unroll
  for (int i = 0; i < 2; i++)
#pragma unroll
    for (int j = 0; j < 4; j++) acc[i][j] = zero8();

  g2l_b128(wt + (size_t)bcol * CH + bofs, sb[0] + bcol * 32 + bofs);

  for (int kb = 0; kb < CH; kb += 32) {
    const int cur = (kb >> 5) & 1;
    const int kbn = (kb + 32) & (CH - 1);
    g2l_b128(wt + (size_t)bcol * CH + kbn + bofs, sb[cur ^ 1] + bcol * 32 + bofs);
    wait_async<1>();
    __syncthreads();

    V16 a[2];
#pragma unroll
    for (int i = 0; i < 2; i++) {
      const _Float16* ar = yh + (size_t)(tm * 32 + i * 16 + row) * CH + kb;
      a[i].h[0] = *(const v8h*)(ar + hi * 16);
      a[i].h[1] = *(const v8h*)(ar + hi * 16 + 8);
      __builtin_prefetch(ar + 64);
    }
    const _Float16* sbc = sb[cur];
#pragma unroll
    for (int j = 0; j < 4; j++) {
      v16h bb = *(const v16h*)(sbc + (j * 16 + row) * 32 + hi * 16);
      acc[0][j] = wmma_f16(a[0].v, bb, acc[0][j]);
      acc[1][j] = wmma_f16(a[1].v, bb, acc[1][j]);
    }
    __syncthreads();
  }

#pragma unroll
  for (int i = 0; i < 2; i++) {
#pragma unroll
    for (int j = 0; j < 4; j++) {
      const int n    = tn * 64 + j * 16 + row;
      const float bs = b_proj[n];
#pragma unroll
      for (int vv = 0; vv < 8; vv++) {
        const int tok = tm * 32 + i * 16 + vv + hi * 8;
        out[(size_t)tok * CH + n] = acc[i][j][vv] + bs;
      }
    }
  }
}

// ---------------------------------------------------------------------------
// Launcher
// ---------------------------------------------------------------------------
extern "C" void kernel_launch(void* const* d_in, const int* in_sizes, int n_in,
                              void* d_out, int out_size, void* d_ws,
                              size_t ws_size, hipStream_t stream) {
  const float* x      = (const float*)d_in[0];
  const float* w_attn = (const float*)d_in[1];
  const float* b_attn = (const float*)d_in[2];
  const float* w_proj = (const float*)d_in[3];
  const float* b_proj = (const float*)d_in[4];
  float* out = (float*)d_out;

  // Workspace partition (all offsets are >=2MB multiples -> aligned).
  char* ws = (char*)d_ws;
  _Float16* xh  = (_Float16*)(ws + 0);                 // 16 MiB  x in f16
  _Float16* waT = (_Float16*)(ws + 16777216);          // 6 MiB   w_attn^T f16
  _Float16* wpT = (_Float16*)(ws + 23068672);          // 2 MiB   w_proj^T f16
  _Float16* qh  = (_Float16*)(ws + 25165824);          // 16 MiB  q [B,H,T,D]
  _Float16* kh  = (_Float16*)(ws + 41943040);          // 16 MiB  k [B,H,T,D]
  _Float16* vTh = (_Float16*)(ws + 58720256);          // 16 MiB  v [B,H,D,T]
  _Float16* yh  = (_Float16*)(ws + 75497472);          // 16 MiB  attn out f16

  // 1) precision conversion (f32 -> f16; weights transposed for B-operand)
  cvt_f32_to_f16<<<(NTOK * CH) / 256, 256, 0, stream>>>(x, xh, NTOK * CH);
  cvt_f32_to_f16_T<<<(CH * N3C) / 256, 256, 0, stream>>>(w_attn, waT, CH, N3C);
  cvt_f32_to_f16_T<<<(CH * CH) / 256, 256, 0, stream>>>(w_proj, wpT, CH, CH);

  // 2) QKV GEMM: 1536 blocks x 8 waves; blocks grouped (32 M-groups x 48 N-tiles)
  qkv_gemm<<<1536, 256, 0, stream>>>(xh, waT, b_attn, qh, kh, vTh);

  // 3) flash attention (8192 waves = 1024 blocks x 8 waves)
  flash_attn<<<1024, 256, 0, stream>>>(qh, kh, vTh, yh);

  // 4) output projection: 512 blocks (32 M-groups x 16 N-tiles)
  proj_gemm<<<512, 256, 0, stream>>>(yh, wpT, b_proj, out);
}
